// BlockDirectTwice_8546984919218
// MI455X (gfx1250) — matrix operand order repaired and checked by reading, not compiled, not run on hardware
//
#include <hip/hip_runtime.h>
#include <math.h>

// ---------------------------------------------------------------------------
// Dual-branch transformer block for MI455X (gfx1250, wave32, WMMA + TDM).
// GEMMs: v_wmma_f32_16x16x32_f16, tiles staged via tensor_load_to_lds (TDM,
// TENSORcnt double-buffered). Attention: K/V tiles TDM-staged to LDS, shared
// by the 4 waves of a block. Softmax reductions: DPP16 row ops (no LDS).
// ---------------------------------------------------------------------------

typedef _Float16 half_t;
typedef __attribute__((ext_vector_type(8)))  _Float16 v8h;
typedef __attribute__((ext_vector_type(16))) _Float16 v16h;
typedef __attribute__((ext_vector_type(8)))  float    v8f;

#define D_MODEL 768
#define N_HEAD  12
#define HEAD_DIM 64
#define SEQ     1024
#define BATCH   8
#define NTOK    (BATCH*SEQ)      // 8192
#define DFF_    3072

#if defined(__has_builtin)
#if __has_builtin(__builtin_amdgcn_tensor_load_to_lds)
#define HAVE_TDM 1
#endif
#endif
#ifndef HAVE_TDM
#define HAVE_TDM 0
#endif

__device__ __forceinline__ v8f wmma16(v16h a, v16h b, v8f c) {
  // D = A(16x32 f16) * B(32x16 f16) + C(16x16 f32)
  return __builtin_amdgcn_wmma_f32_16x16x32_f16(false, a, false, b, (short)0, c,
                                                false, false);
}
__device__ __forceinline__ v16h mk16(v8h lo, v8h hi) {
  return __builtin_shufflevector(lo, hi, 0,1,2,3,4,5,6,7,8,9,10,11,12,13,14,15);
}

// --- DPP16 16-lane reductions (VALU only, no ds_bpermute round trips) ------
template<int CTRL>
__device__ __forceinline__ float dppf(float x) {
  return __int_as_float(
      __builtin_amdgcn_update_dpp(0, __float_as_int(x), CTRL, 0xF, 0xF, true));
}
__device__ __forceinline__ float rowmax16(float v) {
  v = fmaxf(v, dppf<0xB1>(v));    // quad_perm(1,0,3,2)  : xor 1
  v = fmaxf(v, dppf<0x4E>(v));    // quad_perm(2,3,0,1)  : xor 2
  v = fmaxf(v, dppf<0x141>(v));   // row_half_mirror     : combine quads in 8
  v = fmaxf(v, dppf<0x140>(v));   // row_mirror          : combine 8s in 16
  return v;
}
__device__ __forceinline__ float rowsum16(float v) {
  v += dppf<0xB1>(v);
  v += dppf<0x4E>(v);
  v += dppf<0x141>(v);
  v += dppf<0x140>(v);
  return v;
}

// ---------------------------------------------------------------------------
// TDM: 2D tile global -> LDS via Tensor DMA descriptor (ISA 08 §8.3/8.4).
// pad_enable inserts LDS padding so tile rows land at a conflict-free pitch.
// ---------------------------------------------------------------------------
#if HAVE_TDM
typedef __attribute__((ext_vector_type(4))) unsigned int u32x4;
typedef __attribute__((ext_vector_type(8))) int i32x8;
typedef __attribute__((ext_vector_type(4))) int i32x4;

__device__ __forceinline__ void tdm_load_2d(unsigned lds_byte_addr,
                                            const half_t* g,
                                            unsigned tile_w /*elems*/,
                                            unsigned tile_h /*rows*/,
                                            unsigned row_stride /*elems*/,
                                            unsigned pad_int_code,
                                            unsigned pad_amt_code) {
  unsigned long long ga = (unsigned long long)(uintptr_t)g;
  u32x4 g0;
  g0.x = 1u;                                                 // count=1 (user)
  g0.y = lds_byte_addr;                                      // lds_addr
  g0.z = (unsigned)ga;                                       // global_addr lo
  g0.w = (unsigned)((ga >> 32) & 0x01FFFFFFull) | (2u << 30);// addr hi | type=2
  const unsigned td0 = row_stride;      // tensor dim0 (no OOB for our tiles)
  const unsigned td1 = 1u << 20;        // tensor dim1: huge
  i32x8 g1;
  g1[0] = (int)((1u << 16) |                        // data_size = 2 bytes
                (1u << 20) |                        // pad_enable
                (pad_int_code << 22) | (pad_amt_code << 25));
  g1[1] = (int)((td0 & 0xFFFFu) << 16);             // barrier=0 | td0 lo16
  g1[2] = (int)(((td0 >> 16) & 0xFFFFu) | ((td1 & 0xFFFFu) << 16));
  g1[3] = (int)(((td1 >> 16) & 0xFFFFu) | ((tile_w & 0xFFFFu) << 16));
  g1[4] = (int)(tile_h & 0xFFFFu);                  // tile_dim1 | tile_dim2=0
  g1[5] = (int)td0;                                 // tensor_dim0_stride lo32
  g1[6] = 0;
  g1[7] = 0;
  const i32x4 z4 = (i32x4){0, 0, 0, 0};
#if __clang_major__ >= 23
  const i32x8 z8 = (i32x8){0, 0, 0, 0, 0, 0, 0, 0};
  __builtin_amdgcn_tensor_load_to_lds(g0, g1, z4, z4, z8, 0);
#else
  __builtin_amdgcn_tensor_load_to_lds(g0, g1, z4, z4, 0);
#endif
}
#endif  // HAVE_TDM

// ---------------------------------------------------------------------------
// fp32 -> f16 conversion (weights, once per launch)
// ---------------------------------------------------------------------------
__global__ void cvt_f32_f16(const float* __restrict__ in, half_t* __restrict__ out,
                            size_t n) {
  size_t i  = (size_t)blockIdx.x * blockDim.x + threadIdx.x;
  size_t st = (size_t)gridDim.x * blockDim.x;
  for (; i < n; i += st) out[i] = (half_t)in[i];
}

// ---------------------------------------------------------------------------
// LayerNorm over D=768, one block (256 thr) per token, f16 output
// ---------------------------------------------------------------------------
__global__ __launch_bounds__(256) void ln_f16(const float* __restrict__ x,
                                              const float* __restrict__ g,
                                              const float* __restrict__ b,
                                              half_t* __restrict__ out) {
  __shared__ float red[256];
  const int t = blockIdx.x;
  const float* xr = x + (size_t)t * D_MODEL;
  float s = 0.f, s2 = 0.f;
  for (int i = threadIdx.x; i < D_MODEL; i += 256) {
    float v = xr[i]; s += v; s2 += v * v;
  }
  red[threadIdx.x] = s; __syncthreads();
  for (int o = 128; o > 0; o >>= 1) {
    if (threadIdx.x < o) red[threadIdx.x] += red[threadIdx.x + o];
    __syncthreads();
  }
  const float mu = red[0] * (1.0f / D_MODEL);
  __syncthreads();
  red[threadIdx.x] = s2; __syncthreads();
  for (int o = 128; o > 0; o >>= 1) {
    if (threadIdx.x < o) red[threadIdx.x] += red[threadIdx.x + o];
    __syncthreads();
  }
  const float var = red[0] * (1.0f / D_MODEL) - mu * mu;
  const float rs  = rsqrtf(var + 1e-6f);
  for (int i = threadIdx.x; i < D_MODEL; i += 256)
    out[(size_t)t * D_MODEL + i] = (half_t)((xr[i] - mu) * rs * g[i] + b[i]);
}

// ---------------------------------------------------------------------------
// Weighted combine: hA = 0.6*hA + 0.4*hB
// ---------------------------------------------------------------------------
__global__ void combine_k(float* __restrict__ hA, const float* __restrict__ hB,
                          size_t n) {
  size_t i  = (size_t)blockIdx.x * blockDim.x + threadIdx.x;
  size_t st = (size_t)gridDim.x * blockDim.x;
  for (; i < n; i += st) hA[i] = 0.6f * hA[i] + 0.4f * hB[i];
}

// ---------------------------------------------------------------------------
// Tiled WMMA GEMM: out = epilogue(A[M,K](f16) @ Bw[K,N](f16) + bias)
//   block: 256 thr = 8 waves, tile 128x128, BK=32, wave tile 32x64.
//   TDM path: wave0 drives a double-buffered tensor_load_to_lds pipeline.
// ---------------------------------------------------------------------------
#define BM 128
#define BN 128
#define BKK 32
#define ASTR 40    // 32 + 8 halves pad  (row = 16 dwords + 4 dwords pad)
#define BSTR 136   // 128 + 8 halves pad (row = 64 dwords + 4 dwords pad)

template<bool GELU_ACT, bool OUT_HALF, bool RESID>
__global__ __launch_bounds__(256) void gemm_f16(
    const half_t* __restrict__ A, const half_t* __restrict__ Bw,
    const float* __restrict__ bias, const float* __restrict__ resid,
    void* __restrict__ outp, int M, int N, int K) {
  const int tid  = threadIdx.x;
  const int wave = tid >> 5, lane = tid & 31;
  const int lh   = lane >> 4, lc = lane & 15;
  const int bm   = blockIdx.y * BM, bn = blockIdx.x * BN;
  const int wrow = (wave & 3) * 32, wcol = (wave >> 2) * 64;

  v8f acc[2][4];
  #pragma unroll
  for (int i = 0; i < 2; i++)
    #pragma unroll
    for (int j = 0; j < 4; j++) acc[i][j] = (v8f){0,0,0,0,0,0,0,0};

#if HAVE_TDM
  __shared__ half_t As[2][BM * ASTR];
  __shared__ half_t Bs[2][BKK * BSTR];
  const int nk = K / BKK;
  if (wave == 0) {   // prologue: tile 0 in flight
    tdm_load_2d((unsigned)(uintptr_t)&As[0][0], A + (size_t)bm * K,
                BKK, BM, K, /*16dw*/3, /*4dw*/3);
    tdm_load_2d((unsigned)(uintptr_t)&Bs[0][0], Bw + bn,
                BN, BKK, N, /*64dw*/5, /*4dw*/3);
  }
  for (int it = 0; it < nk; it++) {
    if (wave == 0) {
      if (it + 1 < nk) {
        tdm_load_2d((unsigned)(uintptr_t)&As[(it + 1) & 1][0],
                    A + (size_t)bm * K + (size_t)(it + 1) * BKK,
                    BKK, BM, K, 3, 3);
        tdm_load_2d((unsigned)(uintptr_t)&Bs[(it + 1) & 1][0],
                    Bw + (size_t)(it + 1) * BKK * N + bn,
                    BN, BKK, N, 5, 3);
        __builtin_amdgcn_s_wait_tensorcnt((short)2);  // tile `it` landed
      } else {
        __builtin_amdgcn_s_wait_tensorcnt((short)0);
      }
    }
    __syncthreads();
    const half_t* Asb = &As[it & 1][0];
    const half_t* Bsb = &Bs[it & 1][0];
#else
  __shared__ half_t AsBuf[BM * ASTR];
  __shared__ half_t BsBuf[BKK * BSTR];
  const int arow = tid >> 1, aseg = (tid & 1) * 16;  // 128 rows x 2 x 16 halves
  const int brow = tid >> 3, bn0  = (tid & 7) * 16;  // 32 k-rows x 8 x 16 n
  const int nk = K / BKK;
  for (int it = 0; it < nk; it++) {
    const int k0 = it * BKK;
    const half_t* ga = A + (size_t)(bm + arow) * K + k0 + aseg;
    v8h a0 = *(const v8h*)ga;
    v8h a1 = *(const v8h*)(ga + 8);
    const half_t* gb = Bw + (size_t)(k0 + brow) * N + bn + bn0;
    v8h b0 = *(const v8h*)gb;
    v8h b1 = *(const v8h*)(gb + 8);
    *(v8h*)(AsBuf + arow * ASTR + aseg)     = a0;
    *(v8h*)(AsBuf + arow * ASTR + aseg + 8) = a1;
    *(v8h*)(BsBuf + brow * BSTR + bn0)      = b0;
    *(v8h*)(BsBuf + brow * BSTR + bn0 + 8)  = b1;
    __syncthreads();
    const half_t* Asb = AsBuf;
    const half_t* Bsb = BsBuf;
#endif
    // A frags (vectorized): row = lc, K = e + 16*lh (contiguous)
    v16h af[2];
    #pragma unroll
    for (int i = 0; i < 2; i++) {
      const half_t* p = Asb + (wrow + i * 16 + lc) * ASTR + lh * 16;
      af[i] = mk16(*(const v8h*)p, *(const v8h*)(p + 8));
    }
    // B frags: col = lc, K = e + 16*lh (strided over [k][n] tile)
    v16h bf[4];
    #pragma unroll
    for (int j = 0; j < 4; j++) {
      const int col = wcol + j * 16 + lc;
      v16h b;
      #pragma unroll
      for (int e = 0; e < 16; e++) b[e] = Bsb[(e + 16 * lh) * BSTR + col];
      bf[j] = b;
    }
    #pragma unroll
    for (int i = 0; i < 2; i++)
      #pragma unroll
      for (int j = 0; j < 4; j++)
        acc[i][j] = wmma16(af[i], bf[j], acc[i][j]);
    __syncthreads();
  }

  // epilogue: C layout -> lane col = lc, row = r + 8*lh
  #pragma unroll
  for (int i = 0; i < 2; i++) {
    #pragma unroll
    for (int j = 0; j < 4; j++) {
      const int gcol = bn + wcol + j * 16 + lc;
      const float bz = bias ? bias[gcol] : 0.0f;
      #pragma unroll
      for (int r = 0; r < 8; r++) {
        const int grow = bm + wrow + i * 16 + r + 8 * lh;
        float v = acc[i][j][r] + bz;
        if (GELU_ACT) v = 0.5f * v * (1.0f + erff(v * 0.70710678118654752f));
        if (RESID)    v += resid[(size_t)grow * N + gcol];
        if (OUT_HALF) ((half_t*)outp)[(size_t)grow * N + gcol] = (half_t)v;
        else          ((float*)outp)[(size_t)grow * N + gcol]  = v;
      }
    }
  }
}

// ---------------------------------------------------------------------------
// Flash attention: 4 waves/block share one (b,h); each wave owns a 16-row
// Q tile; 32 keys/iter. K/V tiles (32x64) are TDM-staged into LDS once per
// block, double-buffered on TENSORcnt. QK^T: 4 WMMA; softmax: DPP16; P via
// LDS transpose; PV: 4 WMMA.
// ---------------------------------------------------------------------------
#define KVSTR 72   // 64 + 8 halves pad (row = 32 dwords + 4 dwords pad)

__global__ __launch_bounds__(128) void attn_k(const half_t* __restrict__ Q,
                                              const half_t* __restrict__ Km,
                                              const half_t* __restrict__ Vm,
                                              half_t* __restrict__ O) {
  __shared__ half_t pl[4 * 16 * 32];
#if HAVE_TDM
  __shared__ half_t KVl[2][2][32 * KVSTR];   // [buf][K=0/V=1][32 rows x 72]
#endif
  const int tid  = threadIdx.x;
  const int wave = tid >> 5, lane = tid & 31;
  const int lh   = lane >> 4, lc = lane & 15;
  const int tile = blockIdx.x * 4 + wave;      // 6144 tiles total
  const int qt   = tile & 63;                  // S/16 = 64; block stays in 1 bh
  const int bh   = tile >> 6;
  const int b    = bh / N_HEAD, h = bh % N_HEAD;
  const size_t qbase  = ((size_t)b * SEQ + qt * 16) * D_MODEL + h * HEAD_DIM;
  const size_t kvbase = (size_t)b * SEQ * D_MODEL + h * HEAD_DIM;
  half_t* myp = pl + wave * (16 * 32);

  // Q fragments (A layout): row = lc; features 0..31 and 32..63
  const half_t* qp = Q + qbase + (size_t)lc * D_MODEL;
  const v16h aq0 = mk16(*(const v8h*)(qp +      16 * lh), *(const v8h*)(qp +      16 * lh + 8));
  const v16h aq1 = mk16(*(const v8h*)(qp + 32 + 16 * lh), *(const v8h*)(qp + 32 + 16 * lh + 8));

  v8f acc[4];
  #pragma unroll
  for (int t = 0; t < 4; t++) acc[t] = (v8f){0,0,0,0,0,0,0,0};
  float m[8], l[8];
  #pragma unroll
  for (int r = 0; r < 8; r++) { m[r] = -3.0e38f; l[r] = 0.f; }

#if HAVE_TDM
  if (wave == 0) {   // prologue: K/V tile 0 in flight
    tdm_load_2d((unsigned)(uintptr_t)&KVl[0][0][0], Km + kvbase,
                HEAD_DIM, 32, D_MODEL, /*32dw*/4, /*4dw*/3);
    tdm_load_2d((unsigned)(uintptr_t)&KVl[0][1][0], Vm + kvbase,
                HEAD_DIM, 32, D_MODEL, 4, 3);
  }
#endif

  for (int kt = 0; kt < SEQ / 32; kt++) {
#if HAVE_TDM
    if (wave == 0) {
      if (kt + 1 < SEQ / 32) {
        const size_t nkb = kvbase + (size_t)(kt + 1) * 32 * D_MODEL;
        tdm_load_2d((unsigned)(uintptr_t)&KVl[(kt + 1) & 1][0][0], Km + nkb,
                    HEAD_DIM, 32, D_MODEL, 4, 3);
        tdm_load_2d((unsigned)(uintptr_t)&KVl[(kt + 1) & 1][1][0], Vm + nkb,
                    HEAD_DIM, 32, D_MODEL, 4, 3);
        __builtin_amdgcn_s_wait_tensorcnt((short)2);  // tile `kt` landed
      } else {
        __builtin_amdgcn_s_wait_tensorcnt((short)0);
      }
    }
    __syncthreads();   // K/V tile `kt` visible to all 4 waves
    const half_t* Kb = &KVl[kt & 1][0][0];
    const half_t* Vb = &KVl[kt & 1][1][0];
    const half_t* kp0 = Kb + (size_t)lc * KVSTR;
    const half_t* kp1 = Kb + (size_t)(16 + lc) * KVSTR;
#else
    const size_t kbg = kvbase + (size_t)kt * 32 * D_MODEL;
    const half_t* kp0 = Km + kbg + (size_t)lc * D_MODEL;
    const half_t* kp1 = Km + kbg + (size_t)(16 + lc) * D_MODEL;
#endif
    // K fragments (B layout): col = key lane, K = feature j + 16*lh
    const v16h bk00 = mk16(*(const v8h*)(kp0 +      16 * lh), *(const v8h*)(kp0 +      16 * lh + 8));
    const v16h bk01 = mk16(*(const v8h*)(kp0 + 32 + 16 * lh), *(const v8h*)(kp0 + 32 + 16 * lh + 8));
    const v16h bk10 = mk16(*(const v8h*)(kp1 +      16 * lh), *(const v8h*)(kp1 +      16 * lh + 8));
    const v16h bk11 = mk16(*(const v8h*)(kp1 + 32 + 16 * lh), *(const v8h*)(kp1 + 32 + 16 * lh + 8));

    v8f s0 = (v8f){0,0,0,0,0,0,0,0}, s1 = (v8f){0,0,0,0,0,0,0,0};
    s0 = wmma16(aq0, bk00, s0); s0 = wmma16(aq1, bk01, s0);
    s1 = wmma16(aq0, bk10, s1); s1 = wmma16(aq1, bk11, s1);

    // online softmax over 32 keys; rows live across a 16-lane group
    #pragma unroll
    for (int r = 0; r < 8; r++) {
      float a = s0[r] * 0.125f, c = s1[r] * 0.125f;
      const float mx   = rowmax16(fmaxf(a, c));
      const float mn   = fmaxf(m[r], mx);
      const float corr = __expf(m[r] - mn);
      const float p0   = __expf(a - mn);
      const float p1   = __expf(c - mn);
      const float rs   = rowsum16(p0 + p1);
      l[r] = l[r] * corr + rs;
      m[r] = mn;
      #pragma unroll
      for (int t = 0; t < 4; t++) acc[t][r] *= corr;
      s0[r] = p0; s1[r] = p1;
    }

    __syncthreads();                       // protect WAR on LDS P tile
    #pragma unroll
    for (int r = 0; r < 8; r++) {          // C layout -> LDS row-major 16x32
      myp[(r + 8 * lh) * 32 + lc]      = (half_t)s0[r];
      myp[(r + 8 * lh) * 32 + 16 + lc] = (half_t)s1[r];
    }
    __syncthreads();

    // P as A fragment (16x32), keys are the K dim
    const half_t* pp = myp + lc * 32;
    const v16h ap = mk16(*(const v8h*)(pp + 16 * lh), *(const v8h*)(pp + 16 * lh + 8));

    // V fragments (B layout): col = hd (t*16+lc), K = key j + 16*lh
    #pragma unroll
    for (int t = 0; t < 4; t++) {
      v16h bv;
      #pragma unroll
      for (int j = 0; j < 16; j++)
#if HAVE_TDM
        bv[j] = Vb[(size_t)(j + 16 * lh) * KVSTR + t * 16 + lc];
#else
        bv[j] = Vm[kbg + (size_t)(j + 16 * lh) * D_MODEL + t * 16 + lc];
#endif
      acc[t] = wmma16(ap, bv, acc[t]);
    }
#if HAVE_TDM
    __syncthreads();   // all waves done with buf `kt` before it is re-filled
#endif
  }

  #pragma unroll
  for (int r = 0; r < 8; r++) {
    const float inv = 1.0f / l[r];
    #pragma unroll
    for (int t = 0; t < 4; t++)
      O[qbase + (size_t)(r + 8 * lh) * D_MODEL + t * 16 + lc] =
          (half_t)(acc[t][r] * inv);
  }
}

// ---------------------------------------------------------------------------
// Host orchestration
// ---------------------------------------------------------------------------
extern "C" void kernel_launch(void* const* d_in, const int* in_sizes, int n_in,
                              void* d_out, int out_size, void* d_ws, size_t ws_size,
                              hipStream_t stream) {
  const float* x0    = (const float*)d_in[0];
  const float* x1    = (const float*)d_in[1];
  const float* ln0_g = (const float*)d_in[2];
  const float* ln0_b = (const float*)d_in[3];
  const float* ln1_g = (const float*)d_in[4];
  const float* ln1_b = (const float*)d_in[5];
  const float* lnf_g = (const float*)d_in[6];
  const float* lnf_b = (const float*)d_in[7];
  const float* fc1_w = (const float*)d_in[8];
  const float* fc1_b = (const float*)d_in[9];
  const float* fc2_w = (const float*)d_in[10];
  const float* fc2_b = (const float*)d_in[11];
  const float* aw[2][4]; const float* ab[2][4];
  for (int br = 0; br < 2; br++)
    for (int i = 0; i < 4; i++) {
      aw[br][i] = (const float*)d_in[12 + br * 8 + i];      // wq,wk,wv,wo
      ab[br][i] = (const float*)d_in[12 + br * 8 + 4 + i];  // bq,bk,bv,bo
    }

  const size_t DD  = (size_t)D_MODEL * D_MODEL;   // 589824
  const size_t ND  = (size_t)NTOK * D_MODEL;      // 6291456
  const size_t DFW = (size_t)D_MODEL * DFF_;      // 2359296

  half_t* H = (half_t*)d_ws;   // indexed in halves
  half_t* w16[2][4];
  for (int br = 0; br < 2; br++)
    for (int i = 0; i < 4; i++) w16[br][i] = H + (br * 4 + i) * DD;
  half_t* fc1_16 = H + 8 * DD;
  half_t* fc2_16 = fc1_16 + DFW;
  const size_t xoff = 8 * DD + 2 * DFW;
  half_t* xn0 = H + xoff;           // LN output / later ctx0
  half_t* xn1 = xn0 + ND;           // LN output / later ctx1
  const size_t qoff = xoff + 2 * ND;
  half_t* qkv[2][3];
  for (int br = 0; br < 2; br++)
    for (int i = 0; i < 3; i++) qkv[br][i] = H + qoff + (br * 3 + i) * ND;
  // hA/hB (f32) reuse the dead q/k/v region after attention completes
  float* hA = (float*)(H + qoff);            // ND floats == q0+k0 halves
  float* hB = (float*)(H + qoff + 2 * ND);   // ND floats == v0+q1 halves
  half_t* hn16 = H + qoff + 4 * ND;          // k1 region, free after attention
  half_t* ff1  = H + qoff + 6 * ND;          // [NTOK, DFF] f16

  // 1) weight conversion fp32 -> f16
  for (int br = 0; br < 2; br++)
    for (int i = 0; i < 4; i++)
      cvt_f32_f16<<<1024, 256, 0, stream>>>(aw[br][i], w16[br][i], DD);
  cvt_f32_f16<<<2048, 256, 0, stream>>>(fc1_w, fc1_16, DFW);
  cvt_f32_f16<<<2048, 256, 0, stream>>>(fc2_w, fc2_16, DFW);

  // 2) pre-attention LayerNorms -> f16
  ln_f16<<<NTOK, 256, 0, stream>>>(x0, ln0_g, ln0_b, xn0);
  ln_f16<<<NTOK, 256, 0, stream>>>(x1, ln1_g, ln1_b, xn1);

  // 3) QKV projections (f16 out)
  const dim3 gD(D_MODEL / BN, NTOK / BM);
  for (int br = 0; br < 2; br++) {
    half_t* src = br ? xn1 : xn0;
    for (int i = 0; i < 3; i++)
      gemm_f16<false, true, false><<<gD, 256, 0, stream>>>(
          src, w16[br][i], ab[br][i], nullptr, (void*)qkv[br][i],
          NTOK, D_MODEL, D_MODEL);
  }

  // 4) flash attention, ctx overwrites LN buffers
  attn_k<<<1536, 128, 0, stream>>>(qkv[0][0], qkv[0][1], qkv[0][2], xn0);
  attn_k<<<1536, 128, 0, stream>>>(qkv[1][0], qkv[1][1], qkv[1][2], xn1);

  // 5) output projections + residual (f32 out)
  gemm_f16<false, false, true><<<gD, 256, 0, stream>>>(
      xn0, w16[0][3], ab[0][3], x0, (void*)hA, NTOK, D_MODEL, D_MODEL);
  gemm_f16<false, false, true><<<gD, 256, 0, stream>>>(
      xn1, w16[1][3], ab[1][3], x1, (void*)hB, NTOK, D_MODEL, D_MODEL);

  // 6) weighted combine: hA = 0.6*hA + 0.4*hB
  combine_k<<<4096, 256, 0, stream>>>(hA, hB, ND);

  // 7) final LN -> f16
  ln_f16<<<NTOK, 256, 0, stream>>>(hA, lnf_g, lnf_b, hn16);

  // 8) fc1 + exact GELU (f16 out)
  gemm_f16<true, true, false><<<dim3(DFF_ / BN, NTOK / BM), 256, 0, stream>>>(
      hn16, fc1_16, fc1_b, nullptr, (void*)ff1, NTOK, DFF_, D_MODEL);

  // 9) fc2 + bias + h residual -> d_out (f32)
  gemm_f16<false, false, true><<<gD, 256, 0, stream>>>(
      ff1, fc2_16, fc2_b, hA, d_out, NTOK, D_MODEL, DFF_);
}
